// LSTMlayer_90821378441596
// MI455X (gfx1250) — matrix operand order repaired and checked by reading, not compiled, not run on hardware
//
#include <hip/hip_runtime.h>
#include <math.h>

// Problem constants (match reference)
#define Bv 64
#define Tv 512
#define Dv 512
#define Hv 1024
#define G4H (4 * Hv)          // 4096
#define KTOT (Dv + Hv)        // 1536 combined K
#define KT_X (Dv / 32)        // 16 k-tiles from x
#define KT_H (Hv / 32)        // 32 k-tiles from h
#define KT_ALL (KTOT / 32)    // 48 k-tiles total
#define NTILES (G4H / 16)     // 256 column tiles of 16
#define KT_STRIDE (NTILES * 512)   // elements per ktile row of packed W (131072)
#define GATE_STRIDE (64 * 512)     // elements between gate tiles, same kt (32768)

typedef __attribute__((ext_vector_type(16))) __bf16 v16bf;
typedef __attribute__((ext_vector_type(8)))  __bf16 v8bf;
typedef __attribute__((ext_vector_type(8)))  float  v8f;

__device__ __forceinline__ unsigned short f2bf(float f) {
    union { float f; unsigned u; } v; v.f = f;
    unsigned r = v.u + 0x7FFFu + ((v.u >> 16) & 1u);   // round-to-nearest-even
    return (unsigned short)(r >> 16);
}

__device__ __forceinline__ float fast_sigmoid(float x) {
    return 1.0f / (1.0f + __expf(-x));
}
// tanh(x) = 1 - 2/(e^{2x}+1); saturates correctly at +/-inf, branch-free
__device__ __forceinline__ float fast_tanh(float x) {
    float e = __expf(2.0f * x);
    return 1.0f - 2.0f / (e + 1.0f);
}

// Assemble a WMMA A-operand from two contiguous 16B bf16 chunks
// (ap must already include the lane-dependent K sub-chunk offset).
__device__ __forceinline__ v16bf load_a_tile(const unsigned short* ap) {
    v8bf lo = *(const v8bf*)(ap);        // K = off .. off+7
    v8bf hi = *(const v8bf*)(ap + 8);    // K = 8+off .. 15+off
    v16bf a;
#pragma unroll
    for (int i = 0; i < 8; ++i) { a[i] = lo[i]; a[8 + i] = hi[i]; }
    return a;
}

// ---------------------------------------------------------------------------
// Convert x [B,T,D] f32 -> bf16 (row-major, A-matrix friendly)
// ---------------------------------------------------------------------------
__global__ __launch_bounds__(256) void conv_x_kernel(const float* __restrict__ x,
                                                     unsigned short* __restrict__ xbf,
                                                     int n) {
    int i = (blockIdx.x * blockDim.x + threadIdx.x) * 4;
    if (i >= n) return;
    float4 v = *(const float4*)(x + i);
    xbf[i + 0] = f2bf(v.x);
    xbf[i + 1] = f2bf(v.y);
    xbf[i + 2] = f2bf(v.z);
    xbf[i + 3] = f2bf(v.w);
}

// ---------------------------------------------------------------------------
// Pack W = [Wx; Wh]  ([1536,4096] f32, row-major) into WMMA-B tile layout:
// tile(kt,nt) is 32 lanes x 16 bf16 contiguous; lane<16 holds K=kt*32+j of
// column nt*16+lane, lane>=16 holds K=kt*32+16+j of column nt*16+(lane-16).
// ---------------------------------------------------------------------------
__global__ __launch_bounds__(256) void pack_w_kernel(const float* __restrict__ Wx,
                                                     const float* __restrict__ Wh,
                                                     unsigned short* __restrict__ Wpk) {
    int gid = blockIdx.x * blockDim.x + threadIdx.x;    // tile*32 + lane
    if (gid >= KT_ALL * NTILES * 32) return;
    int lane = gid & 31;
    int tile = gid >> 5;
    int nt = tile % NTILES;
    int kt = tile / NTILES;
    int col = nt * 16 + (lane & 15);
    int kbase = kt * 32 + ((lane >> 4) ? 16 : 0);
    unsigned short* dst = Wpk + (size_t)tile * 512 + lane * 16;
#pragma unroll
    for (int j = 0; j < 16; ++j) {
        int k = kbase + j;
        float v = (k < Dv) ? Wx[(size_t)k * G4H + col]
                           : Wh[(size_t)(k - Dv) * G4H + col];
        dst[j] = f2bf(v);
    }
}

// ---------------------------------------------------------------------------
// Zero-init c (f32) and h0 (bf16)
// ---------------------------------------------------------------------------
__global__ __launch_bounds__(256) void init_state_kernel(float* __restrict__ c,
                                                         unsigned short* __restrict__ h0,
                                                         int n) {
    int i = blockIdx.x * blockDim.x + threadIdx.x;
    if (i < n) { c[i] = 0.0f; h0[i] = 0; }
}

// ---------------------------------------------------------------------------
// One LSTM timestep, fully fused:
//   z = [x_t, h_prev] @ [Wx; Wh] + b   (bf16 WMMA, f32 accumulate)
//   i,f,g,o gates -> c,h update -> out[:,t,:], h_next(bf16)
// grid.x = H/16; 4 wave32s per block; each wave owns 16 batch rows and the
// four gate accumulator tiles for this block's 16 h-columns.
// Register double-buffered K pipeline: ktile k+1 loads issue before ktile k's
// WMMAs so VMEM latency hides under matrix math.
// ---------------------------------------------------------------------------
__global__ __launch_bounds__(128) void lstm_step_kernel(
        const unsigned short* __restrict__ xbf,
        const unsigned short* __restrict__ Wpk,
        const float* __restrict__ bias,
        const unsigned short* __restrict__ hprev,
        unsigned short* __restrict__ hnext,
        float* __restrict__ c,
        float* __restrict__ out,
        int t) {
    const int lane   = threadIdx.x & 31;
    const int wv     = threadIdx.x >> 5;   // wave 0..3
    const int m0     = wv * 16;            // batch row base
    const int n0     = blockIdx.x * 16;    // per-gate column base
    const int colL   = lane & 15;
    const int laneHi = lane >> 4;          // 0/1
    const int a_off  = laneHi * 16;        // A-tile K sub-chunk select

    // Accumulators initialized with per-column bias (broadcast over 8 rows)
    v8f acc[4];
#pragma unroll
    for (int g = 0; g < 4; ++g) {
        float bvv = bias[g * Hv + n0 + colL];
#pragma unroll
        for (int r = 0; r < 8; ++r) acc[g][r] = bvv;
    }

    const int arow = m0 + colL;            // this lane's A row (batch index)
    const unsigned short* xb = xbf   + ((size_t)arow * Tv + t) * Dv + a_off;
    const unsigned short* hb = hprev + (size_t)arow * Hv + a_off;
    // packed-W pointer at (kt=0, gate=0, ntile=n0/16), this lane's 32B chunk
    const unsigned short* bb = Wpk + (size_t)n0 * 32 + lane * 16;

    // ---- prologue: ktile 0 ----
    v16bf a_cur = load_a_tile(xb);
    v16bf b_cur[4];
#pragma unroll
    for (int g = 0; g < 4; ++g)
        b_cur[g] = *(const v16bf*)(bb + g * GATE_STRIDE);

    // ---- phase 1: x contribution, ktiles 0..15 (fully unrolled) ----
#pragma unroll
    for (int kt = 0; kt < KT_X; ++kt) {
        const unsigned short* apn = (kt + 1 < KT_X) ? (xb + (kt + 1) * 32) : hb;
        const unsigned short* bbn = bb + KT_STRIDE;
        v16bf a_nxt = load_a_tile(apn);
        v16bf b_nxt[4];
#pragma unroll
        for (int g = 0; g < 4; ++g)
            b_nxt[g] = *(const v16bf*)(bbn + g * GATE_STRIDE);
#pragma unroll
        for (int g = 0; g < 4; ++g)
            acc[g] = __builtin_amdgcn_wmma_f32_16x16x32_bf16(
                false, a_cur, false, b_cur[g], (short)0, acc[g], false, false);
        a_cur = a_nxt;
#pragma unroll
        for (int g = 0; g < 4; ++g) b_cur[g] = b_nxt[g];
        bb = bbn;
    }

    // ---- phase 2: h contribution, ktiles 16..47 (fully unrolled) ----
#pragma unroll
    for (int kt2 = 0; kt2 < KT_H; ++kt2) {
        const bool last = (kt2 + 1 >= KT_H);
        const unsigned short* apn = hb + (last ? kt2 : kt2 + 1) * 32;
        const unsigned short* bbn = bb + (last ? 0 : KT_STRIDE);
        v16bf a_nxt = load_a_tile(apn);
        v16bf b_nxt[4];
#pragma unroll
        for (int g = 0; g < 4; ++g)
            b_nxt[g] = *(const v16bf*)(bbn + g * GATE_STRIDE);
#pragma unroll
        for (int g = 0; g < 4; ++g)
            acc[g] = __builtin_amdgcn_wmma_f32_16x16x32_bf16(
                false, a_cur, false, b_cur[g], (short)0, acc[g], false, false);
        a_cur = a_nxt;
#pragma unroll
        for (int g = 0; g < 4; ++g) b_cur[g] = b_nxt[g];
        bb = bbn;
    }

    // ---- gates + state update (C/D layout: lane = N col, VGPR r = M row) ----
    const int col   = n0 + colL;
    const int rbase = m0 + laneHi * 8;
#pragma unroll
    for (int r = 0; r < 8; ++r) {
        int row = rbase + r;
        float iv = fast_sigmoid(acc[0][r]);
        float fv = fast_sigmoid(acc[1][r]);
        float gv = fast_tanh(acc[2][r]);
        float ov = fast_sigmoid(acc[3][r]);
        size_t sidx = (size_t)row * Hv + col;
        float cn = fv * c[sidx] + iv * gv;
        c[sidx] = cn;
        float hn = ov * fast_tanh(cn);
        out[((size_t)row * Tv + t) * Hv + col] = hn;
        hnext[sidx] = f2bf(hn);
    }
}

// ---------------------------------------------------------------------------
// Host-side launch
// ---------------------------------------------------------------------------
extern "C" void kernel_launch(void* const* d_in, const int* in_sizes, int n_in,
                              void* d_out, int out_size, void* d_ws, size_t ws_size,
                              hipStream_t stream) {
    const float* x    = (const float*)d_in[0];   // [B,T,D]
    const float* Wx   = (const float*)d_in[1];   // [D,4H]
    const float* Wh   = (const float*)d_in[2];   // [H,4H]
    const float* bias = (const float*)d_in[3];   // [4H]
    float* out = (float*)d_out;                  // [B,T,H]

    // Workspace layout:
    //   xbf : B*T*D bf16  (32 MB) | Wpk : 48*256*512 bf16 (12 MB)
    //   h0,h1 : B*H bf16 each     | c : B*H f32
    unsigned short* xbf = (unsigned short*)d_ws;
    unsigned short* Wpk = xbf + (size_t)Bv * Tv * Dv;
    unsigned short* h0  = Wpk + (size_t)KT_ALL * NTILES * 512;
    unsigned short* h1  = h0 + (size_t)Bv * Hv;
    float*          c   = (float*)(h1 + (size_t)Bv * Hv);

    // 1) convert x to bf16
    {
        int n = Bv * Tv * Dv;
        int threads = 256;
        int blocks = (n / 4 + threads - 1) / threads;
        conv_x_kernel<<<blocks, threads, 0, stream>>>(x, xbf, n);
    }
    // 2) pack [Wx;Wh] into WMMA-B tile layout (bf16)
    {
        int n = KT_ALL * NTILES * 32;
        int threads = 256;
        int blocks = (n + threads - 1) / threads;
        pack_w_kernel<<<blocks, threads, 0, stream>>>(Wx, Wh, Wpk);
    }
    // 3) zero h0 / c
    {
        int n = Bv * Hv;
        int threads = 256;
        int blocks = (n + threads - 1) / threads;
        init_state_kernel<<<blocks, threads, 0, stream>>>(c, h0, n);
    }
    // 4) sequential recurrence: one fused kernel per timestep
    unsigned short* hbufs[2] = { h0, h1 };
    for (int t = 0; t < Tv; ++t) {
        unsigned short* hcur = hbufs[t & 1];
        unsigned short* hnxt = hbufs[(t + 1) & 1];
        lstm_step_kernel<<<Hv / 16, 128, 0, stream>>>(
            xbf, Wpk, bias, hcur, hnxt, c, out, t);
    }
}
